// FirstLayer_39659728011898
// MI455X (gfx1250) — matrix operand order repaired and checked by reading, not compiled, not run on hardware
//
#include <hip/hip_runtime.h>
#include <hip/hip_bf16.h>

typedef __bf16 bf16;
typedef __attribute__((ext_vector_type(16))) __bf16 v16bf;
typedef __attribute__((ext_vector_type(8)))  float  v8f;

#define NN    16
#define CIN   32
#define COUT  64
#define BB    16
#define MAXS  64
#define MROWS (NN*MAXS)          /* 1024 padded MLP rows            */
#define SIG2  (0.07f*0.07f)

/* ---------------- workspace layout (bytes) ---------------- */
#define OFF_HDR      0                        /* uint: maxnorm2 bits        */
#define OFF_SLOTHW   256                      /* NN*MAXS int      = 4096    */
#define OFF_COORDS   (OFF_SLOTHW  + 4096)     /* NN*MAXS*2 f32    = 8192    */
#define OFF_ROWVALID (OFF_COORDS  + 8192)     /* MROWS f32        = 4096    */
#define OFF_W0T      (OFF_ROWVALID+ 4096)     /* 128x128 bf16     = 32768   */
#define OFF_W1T      (OFF_W0T     + 32768)
#define OFF_WOUTT    (OFF_W1T     + 32768)    /* 2048x128 bf16    = 524288  */
#define OFF_EMB      (OFF_WOUTT   + 524288)   /* 1024x128 bf16    = 262144  */
#define OFF_H1       (OFF_EMB     + 262144)
#define OFF_H2       (OFF_H1      + 262144)
#define OFF_FT       (OFF_H2      + 262144)   /* 16*64*2048 bf16  = 4194304 */
#define OFF_AG       (OFF_FT      + 4194304)  /* 16*16*2048 bf16  = 1048576 */

/* ---------------- WMMA fragment loaders (ISA 7.12.2) ------ */
__device__ __forceinline__ v16bf load_a_frag(const bf16* p /*rowbase+kb*/, int half) {
    // lanes 0-15: e<8 -> K=e, e>=8 -> K=16+e-8 ; lanes 16-31: +8
    union { v16bf v; uint4 q[2]; } u;
    u.q[0] = *(const uint4*)(p + half*8);
    u.q[1] = *(const uint4*)(p + 16 + half*8);
    return u.v;
}
__device__ __forceinline__ v16bf load_b_frag(const bf16* p /*colbase+kb*/, int half) {
    // lanes 0-15: e<8 -> K=e, e>=8 -> K=16+e-8 ; lanes 16-31: +8
    union { v16bf v; uint4 q[2]; } u;
    const bf16* q = p + half*8;
    u.q[0] = *(const uint4*)(q);
    u.q[1] = *(const uint4*)(q + 16);
    return u.v;
}

/* ---------------- weight transpose+convert ---------------- */
__global__ void convert_weights(const float* __restrict__ W0,
                                const float* __restrict__ W1,
                                const float* __restrict__ Wout,
                                char* __restrict__ ws) {
    int gid = blockIdx.x * 256 + threadIdx.x;   // 294912 total
    if (gid < 16384) {
        int o = gid >> 7, i = gid & 127;
        ((bf16*)(ws + OFF_W0T))[gid] = (bf16)W0[i*128 + o];
    } else if (gid < 32768) {
        int g = gid - 16384; int o = g >> 7, i = g & 127;
        ((bf16*)(ws + OFF_W1T))[g] = (bf16)W1[i*128 + o];
    } else {
        int g = gid - 32768; int o = g >> 7, i = g & 127;
        ((bf16*)(ws + OFF_WOUTT))[g] = (bf16)Wout[i*2048 + o];
    }
}

/* ------ per-n: global max emb-norm^2 + active-site compaction ------ */
__global__ void precompute(const float* __restrict__ v, char* __restrict__ ws) {
    __shared__ float red[256];
    int n = blockIdx.x, tid = threadIdx.x;
    float th = v[n];
    float cs = cosf(th), sn = sinf(th);
    float lmax = 0.f;
    for (int site = tid; site < 4096; site += 256) {
        int h = site >> 6, w = site & 63;
        float x1 = -1.f + h * (2.f/63.f);
        float x2 = -1.f + w * (2.f/63.f);
        float w1 =  cs*x1 + sn*x2;
        float w2 = -sn*x1 + cs*x2;
        float s1 = 0.f, s2 = 0.f;
        for (int e = 0; e < 64; ++e) {
            float mu = e * (1.f/63.f);
            float d1 = w1 - mu, d2 = w2 - mu;
            s1 += expf(-d1*d1*(1.f/SIG2));   // emb^2 = exp(-d^2/sig^2)
            s2 += expf(-d2*d2*(1.f/SIG2));
        }
        lmax = fmaxf(lmax, s1 + s2);
    }
    red[tid] = lmax; __syncthreads();
    for (int s = 128; s > 0; s >>= 1) {
        if (tid < s) red[tid] = fmaxf(red[tid], red[tid + s]);
        __syncthreads();
    }
    if (tid == 0) {
        atomicMax((unsigned*)(ws + OFF_HDR), __float_as_uint(red[0]));
        int*   slot = (int*)  (ws + OFF_SLOTHW)   + n*MAXS;
        float* co   = (float*)(ws + OFF_COORDS)   + n*MAXS*2;
        float* rv   = (float*)(ws + OFF_ROWVALID) + n*MAXS;
        int cnt = 0;
        for (int site = 0; site < 4096 && cnt < MAXS; ++site) {
            int h = site >> 6, w = site & 63;
            float x1 = -1.f + h * (2.f/63.f);
            float x2 = -1.f + w * (2.f/63.f);
            float w1 =  cs*x1 + sn*x2;
            float w2 = -sn*x1 + cs*x2;
            if (w1 >= -0.1f && w1 <= 0.1f && w2 >= -0.1f && w2 <= 0.1f) {
                slot[cnt] = site; co[cnt*2] = w1; co[cnt*2+1] = w2; rv[cnt] = 1.f; ++cnt;
            }
        }
        for (int j = cnt; j < MAXS; ++j) { slot[j] = -1; co[j*2] = 0.f; co[j*2+1] = 0.f; rv[j] = 0.f; }
    }
}

/* ------ normalized RBF embedding at active (padded) sites ------ */
__global__ void emb_kernel(char* __restrict__ ws) {
    int gid  = blockIdx.x * 256 + threadIdx.x;   // 1024*128
    int row  = gid >> 7, j = gid & 127;
    int n    = row >> 6, slot = row & 63;
    float x  = ((const float*)(ws + OFF_COORDS))[(n*MAXS + slot)*2 + (j >> 6)];
    float mu = (float)(j & 63) * (1.f/63.f);
    float mn2 = __uint_as_float(*(const unsigned*)(ws + OFF_HDR));
    float inv = rsqrtf(mn2);
    float d   = x - mu;
    float val = expf(-0.5f * d*d * (1.f/SIG2)) * inv;
    ((bf16*)(ws + OFF_EMB))[gid] = (bf16)val;
}

/* ------ gather data columns for the final GEMM ------ */
__global__ void gather_kernel(const float* __restrict__ data, char* __restrict__ ws) {
    int gid = blockIdx.x * 256 + threadIdx.x;    // (n*16+b)*2048 + k
    int k   = gid & 2047;
    int nb  = gid >> 11;
    int b   = nb & 15, n = nb >> 4;
    int s   = k >> 5, c = k & 31;
    int hw  = ((const int*)(ws + OFF_SLOTHW))[n*MAXS + s];
    if (hw < 0) hw = 0;                          // padded slot; F_T row is zero anyway
    ((bf16*)(ws + OFF_AG))[gid] = (bf16)data[(b*CIN + c)*4096 + hw];
}

/* ------ [M x K] @ [K x Nn] via BT[Nn x K], bias + SiLU -> bf16 ------ */
__global__ void gemm_silu(const bf16* __restrict__ A, const bf16* __restrict__ BT,
                          const float* __restrict__ bias, bf16* __restrict__ Out,
                          int K, int Nn) {
    int lane = threadIdx.x & 31;
    int gw   = blockIdx.x * (blockDim.x >> 5) + (threadIdx.x >> 5);
    int nT   = gw % (Nn >> 4), mT = gw / (Nn >> 4);
    int half = lane >> 4, l16 = lane & 15;
    const bf16* arow = A  + (mT*16 + l16) * K;
    const bf16* bcol = BT + (nT*16 + l16) * K;
    v8f acc = {};
    for (int kb = 0; kb < K; kb += 32) {
        v16bf a = load_a_frag(arow + kb, half);
        v16bf b = load_b_frag(bcol + kb, half);
        acc = __builtin_amdgcn_wmma_f32_16x16x32_bf16(false, a, false, b, (short)0, acc, false, false);
    }
    int col = nT*16 + l16;
    float bv = bias[col];
#pragma unroll
    for (int r = 0; r < 8; ++r) {
        float x = acc[r] + bv;
        x = x / (1.f + expf(-x));                       // SiLU
        Out[(mT*16 + r + 8*half) * Nn + col] = (bf16)x;
    }
}

/* ------ filt GEMM: h2 @ WoutT + bout, masked, stored transposed ------ */
__global__ void gemm_filt(const bf16* __restrict__ A, const bf16* __restrict__ BT,
                          const float* __restrict__ bias, const float* __restrict__ rowvalid,
                          bf16* __restrict__ FT) {
    const int K = 128, Nn = 2048;
    int lane = threadIdx.x & 31;
    int gw   = blockIdx.x * (blockDim.x >> 5) + (threadIdx.x >> 5);
    int nT   = gw % (Nn >> 4), mT = gw / (Nn >> 4);
    int half = lane >> 4, l16 = lane & 15;
    const bf16* arow = A  + (mT*16 + l16) * K;
    const bf16* bcol = BT + (nT*16 + l16) * K;
    v8f acc = {};
    for (int kb = 0; kb < K; kb += 32) {
        v16bf a = load_a_frag(arow + kb, half);
        v16bf b = load_b_frag(bcol + kb, half);
        acc = __builtin_amdgcn_wmma_f32_16x16x32_bf16(false, a, false, b, (short)0, acc, false, false);
    }
    int ocol = nT*16 + l16;              // c*64 + d
    int c = ocol >> 6, d = ocol & 63;
    float bv = bias[ocol];
#pragma unroll
    for (int r = 0; r < 8; ++r) {
        int orow = mT*16 + r + 8*half;   // n*64 + s
        float x  = (acc[r] + bv) * rowvalid[orow];   // zero padded rows incl. bias
        int n = orow >> 6, s = orow & 63;
        FT[(n*64 + d)*2048 + s*32 + c] = (bf16)x;
    }
}

/* ------ per-n final contraction: [16 x 2048] @ [2048 x 64] -> out[b,n,d] ------ */
__global__ void gemm_out(const bf16* __restrict__ Ag, const bf16* __restrict__ FT,
                         float* __restrict__ out) {
    const int K = 2048;
    int lane = threadIdx.x & 31;
    int gw   = blockIdx.x * (blockDim.x >> 5) + (threadIdx.x >> 5);   // 64 waves
    int n = gw >> 2, dT = gw & 3;
    int half = lane >> 4, l16 = lane & 15;
    const bf16* arow = Ag + (n*16 + l16) * K;            // row = batch b
    const bf16* bcol = FT + (n*64 + dT*16 + l16) * K;    // col = d
    v8f acc = {};
    for (int kb = 0; kb < K; kb += 32) {
        v16bf a = load_a_frag(arow + kb, half);
        v16bf b = load_b_frag(bcol + kb, half);
        acc = __builtin_amdgcn_wmma_f32_16x16x32_bf16(false, a, false, b, (short)0, acc, false, false);
    }
    int d = dT*16 + l16;
#pragma unroll
    for (int r = 0; r < 8; ++r) {
        int b = r + 8*half;
        out[(b*NN + n)*COUT + d] = acc[r];
    }
}

extern "C" void kernel_launch(void* const* d_in, const int* in_sizes, int n_in,
                              void* d_out, int out_size, void* d_ws, size_t ws_size,
                              hipStream_t stream) {
    const float* data = (const float*)d_in[0];
    const float* v    = (const float*)d_in[1];
    const float* W0   = (const float*)d_in[2];
    const float* b0   = (const float*)d_in[3];
    const float* W1   = (const float*)d_in[4];
    const float* b1   = (const float*)d_in[5];
    const float* Wout = (const float*)d_in[6];
    const float* bout = (const float*)d_in[7];
    float* out = (float*)d_out;
    char*  ws  = (char*)d_ws;

    hipMemsetAsync(ws, 0, 256, stream);                       // maxnorm accumulator
    convert_weights<<<1152, 256, 0, stream>>>(W0, W1, Wout, ws);
    precompute<<<NN, 256, 0, stream>>>(v, ws);
    emb_kernel<<<512, 256, 0, stream>>>(ws);
    gather_kernel<<<2048, 256, 0, stream>>>(data, ws);
    // h1 = silu(emb @ W0 + b0) ; h2 = silu(h1 @ W1 + b1)
    gemm_silu<<<64, 256, 0, stream>>>((const bf16*)(ws+OFF_EMB), (const bf16*)(ws+OFF_W0T), b0,
                                      (bf16*)(ws+OFF_H1), 128, 128);
    gemm_silu<<<64, 256, 0, stream>>>((const bf16*)(ws+OFF_H1), (const bf16*)(ws+OFF_W1T), b1,
                                      (bf16*)(ws+OFF_H2), 128, 128);
    // filt (masked), stored transposed
    gemm_filt<<<1024, 256, 0, stream>>>((const bf16*)(ws+OFF_H2), (const bf16*)(ws+OFF_WOUTT), bout,
                                        (const float*)(ws+OFF_ROWVALID), (bf16*)(ws+OFF_FT));
    // out[b,n,d]
    gemm_out<<<8, 256, 0, stream>>>((const bf16*)(ws+OFF_AG), (const bf16*)(ws+OFF_FT), out);
}
